// PPOv7Network_16226386444553
// MI455X (gfx1250) — matrix operand order repaired and checked by reading, not compiled, not run on hardware
//
#include <hip/hip_runtime.h>
#include <hip/hip_bf16.h>
#include <math.h>

typedef _Float16 half_t;
typedef __attribute__((ext_vector_type(16))) _Float16 v16h;
typedef __attribute__((ext_vector_type(8)))  float    v8f;

#define B_   8
#define N_   128
#define D_   256
#define L_   64
#define EH_  64
#define NG_  3

// Build a 16-half WMMA operand from two 16-byte chunks.
__device__ __forceinline__ v16h mk_frag(const half_t* p0, const half_t* p1) {
  struct P { uint4 a, b; };
  P u;
  u.a = *reinterpret_cast<const uint4*>(p0);
  u.b = *reinterpret_cast<const uint4*>(p1);
  v16h r;
  __builtin_memcpy(&r, &u, sizeof(r));
  return r;
}

// dst[n*K + k] = (half) src[k*N + n]   (weight transpose + f32->f16)
__global__ void k_transpose_h(const float* __restrict__ src, half_t* __restrict__ dst,
                              int K, int N) {
  int idx = blockIdx.x * blockDim.x + threadIdx.x;
  if (idx < K * N) {
    int n = idx / K, k = idx % K;
    dst[idx] = (half_t)src[(size_t)k * N + n];
  }
}

// Conv1d(1,32,k=4,s=2) + ReLU + flatten + Linear -> layer_embed (B,D)
__global__ void k_layer_embed(const float* __restrict__ lc, const float* __restrict__ cw,
                              const float* __restrict__ cb, const float* __restrict__ lpw,
                              const float* __restrict__ lpb, float* __restrict__ le_out) {
  int b = blockIdx.x, t = threadIdx.x;
  __shared__ float lc_s[L_];
  __shared__ float le_s[32 * 31];
  if (t < L_) lc_s[t] = lc[b * L_ + t];
  __syncthreads();
  for (int e = t; e < 32 * 31; e += 256) {
    int c = e / 31, tt = e % 31;
    float s = cb[c];
#pragma unroll
    for (int k = 0; k < 4; ++k) s += lc_s[2 * tt + k] * cw[c * 4 + k];
    le_s[e] = fmaxf(s, 0.f);
  }
  __syncthreads();
  float s = lpb[t];
  for (int e = 0; e < 32 * 31; ++e) s += le_s[e] * lpw[(size_t)e * D_ + t];
  le_out[b * D_ + t] = s;
}

// h = node_feats @ p_node_w + p_node_b ; also f16 copy
__global__ void k_node_proj(const float* __restrict__ nf, const float* __restrict__ pw,
                            const float* __restrict__ pb, float* __restrict__ h,
                            half_t* __restrict__ h16) {
  int r = blockIdx.x, t = threadIdx.x;
  __shared__ float f[11];
  if (t < 11) f[t] = nf[(size_t)r * 11 + t];
  __syncthreads();
  float s = pb[t];
#pragma unroll
  for (int k = 0; k < 11; ++k) s += f[k] * pw[k * D_ + t];
  h[(size_t)r * D_ + t] = s;
  h16[(size_t)r * D_ + t] = (half_t)s;
}

// Generic WMMA GEMM: out = act(A(MxK,f16) @ Wt(NxK,f16)^T + bias)
// grid = (M/16, N/16), block = 32 (one wave per 16x16 tile)
__global__ void k_gemm(const half_t* __restrict__ A, const half_t* __restrict__ Wt,
                       const float* __restrict__ bias, float* __restrict__ oF,
                       half_t* __restrict__ oH, int K, int N, int act) {
  int mt = blockIdx.x, nt = blockIdx.y;
  int lane = threadIdx.x;
  int ln15 = lane & 15;
  int row = mt * 16 + ln15;
  int col = nt * 16 + ln15;
  int abase = (lane < 16) ? 0 : 8;
  int bbase = (lane < 16) ? 0 : 8;
  const half_t* arow = A + (size_t)row * K;
  const half_t* brow = Wt + (size_t)col * K;
  v8f acc = {};
  for (int kk = 0; kk < K; kk += 32) {
    v16h a  = mk_frag(arow + kk + abase, arow + kk + abase + 16);
    v16h bf = mk_frag(brow + kk + bbase, brow + kk + bbase + 16);
    acc = __builtin_amdgcn_wmma_f32_16x16x32_f16(false, a, false, bf, (short)0, acc,
                                                 false, false);
  }
  float bb = bias[col];
  int rofs = (lane & 16) ? 8 : 0;
#pragma unroll
  for (int r = 0; r < 8; ++r) {
    int m = mt * 16 + rofs + r;
    float v = acc[r] + bb;
    if (act == 1) v = fmaxf(v, 0.f);
    size_t o = (size_t)m * N + col;
    if (oF) oF[o] = v;
    if (oH) oH[o] = (half_t)v;
  }
}

// Per receiver node (b,i): T = relu(pre1 + ef@gw1_edge); G = sigmoid(T@gw2 + gb2);
// msg = G * h_j; masked mean/max over j.  One block per (b,i), 8 waves, 256 WMMAs.
__global__ void k_edge(const float* __restrict__ pre1, const half_t* __restrict__ g2t,
                       const float* __restrict__ gb2, const float* __restrict__ gw1e,
                       const float* __restrict__ ef, const unsigned char* __restrict__ adj,
                       const float* __restrict__ h, float* __restrict__ meanb,
                       float* __restrict__ maxb) {
  int bi = blockIdx.x;          // b*128 + i
  int b = bi >> 7;
  int t = threadIdx.x;
  __shared__ __align__(16) half_t Tsh[N_][EH_ + 8];
  __shared__ float sum_s[D_], max_s[D_];
  __shared__ float ge_s[3][EH_];
  __shared__ float ef_s[N_][3];
  __shared__ float mrow_s[N_];
  __shared__ float gb2_s[D_];
  __shared__ float cnt_s;
  if (t < 192) ge_s[t / EH_][t % EH_] = gw1e[t];
  gb2_s[t] = gb2[t];
  for (int e = t; e < N_ * 3; e += 256) ef_s[e / 3][e % 3] = ef[(size_t)bi * N_ * 3 + e];
  if (t < N_) mrow_s[t] = adj[(size_t)bi * N_ + t] ? 1.f : 0.f;
  __syncthreads();
  for (int e = t; e < N_ * EH_; e += 256) {
    int j = e >> 6, k = e & 63;
    float v = pre1[((size_t)(b * N_ + j)) * EH_ + k]
            + ef_s[j][0] * ge_s[0][k] + ef_s[j][1] * ge_s[1][k] + ef_s[j][2] * ge_s[2][k];
    Tsh[j][k] = (half_t)fmaxf(v, 0.f);
  }
  if (t == 0) {
    float c = 0.f;
    for (int j = 0; j < N_; ++j) c += mrow_s[j];
    cnt_s = fmaxf(c, 1.f);
  }
  __syncthreads();
  int wave = t >> 5, lane = t & 31;
  int ln15 = lane & 15;
  int abase = (lane < 16) ? 0 : 8;
  int bbase = (lane < 16) ? 0 : 8;
  int rofs = (lane & 16) ? 8 : 0;
  for (int nn = 0; nn < 2; ++nn) {
    int nt = wave * 2 + nn;
    int col = nt * 16 + ln15;
    const half_t* brow = g2t + (size_t)col * EH_;
    float sacc = 0.f, macc = -INFINITY;
    for (int mt = 0; mt < 8; ++mt) {
      v8f acc = {};
#pragma unroll
      for (int ks = 0; ks < 2; ++ks) {
        const half_t* ar = &Tsh[mt * 16 + ln15][ks * 32 + abase];
        v16h a  = mk_frag(ar, ar + 16);
        const half_t* br = brow + ks * 32 + bbase;
        v16h bf = mk_frag(br, br + 16);
        acc = __builtin_amdgcn_wmma_f32_16x16x32_f16(false, a, false, bf, (short)0, acc,
                                                     false, false);
      }
#pragma unroll
      for (int r = 0; r < 8; ++r) {
        int j = mt * 16 + rofs + r;
        float g = 1.f / (1.f + expf(-(acc[r] + gb2_s[col])));
        float msg = g * h[((size_t)(b * N_ + j)) * D_ + col];
        if (mrow_s[j] != 0.f) { sacc += msg; macc = fmaxf(macc, msg); }
      }
    }
    sacc += __shfl_xor(sacc, 16);
    macc = fmaxf(macc, __shfl_xor(macc, 16));
    if (lane < 16) { sum_s[col] = sacc; max_s[col] = macc; }
  }
  __syncthreads();
  float mv = max_s[t];
  if (isinf(mv)) mv = 0.f;                 // no neighbors -> 0
  meanb[(size_t)bi * D_ + t] = sum_s[t] / cnt_s;
  maxb[(size_t)bi * D_ + t] = mv;
}

// ui16 = f16 concat [h, agg_mean, agg_max]  (1024 x 768)
__global__ void k_concat(const float* __restrict__ h, const float* __restrict__ mn,
                         const float* __restrict__ mx, half_t* __restrict__ ui16) {
  int idx = blockIdx.x * blockDim.x + threadIdx.x;
  if (idx < B_ * N_ * 3 * D_) {
    int r = idx / (3 * D_), c = idx % (3 * D_);
    float v = (c < D_) ? h[(size_t)r * D_ + c]
            : (c < 2 * D_) ? mn[(size_t)r * D_ + c - D_]
                           : mx[(size_t)r * D_ + c - 2 * D_];
    ui16[idx] = (half_t)v;
  }
}

// h = LN(h + u2out) * g + b  (one block per row)
__global__ void k_ln(const float* __restrict__ u2o, const float* __restrict__ g,
                     const float* __restrict__ bta, float* __restrict__ h,
                     half_t* __restrict__ h16) {
  int r = blockIdx.x, t = threadIdx.x;
  __shared__ float red[D_];
  __shared__ float mu_s, var_s;
  float x = h[(size_t)r * D_ + t] + u2o[(size_t)r * D_ + t];
  red[t] = x;
  __syncthreads();
  for (int s = 128; s > 0; s >>= 1) { if (t < s) red[t] += red[t + s]; __syncthreads(); }
  if (t == 0) mu_s = red[0] / (float)D_;
  __syncthreads();
  float d0 = x - mu_s;
  red[t] = d0 * d0;
  __syncthreads();
  for (int s = 128; s > 0; s >>= 1) { if (t < s) red[t] += red[t + s]; __syncthreads(); }
  if (t == 0) var_s = red[0] / (float)D_;
  __syncthreads();
  float y = d0 * rsqrtf(var_s + 1e-5f) * g[t] + bta[t];
  h[(size_t)r * D_ + t] = y;
  h16[(size_t)r * D_ + t] = (half_t)y;
}

// Heads: graph_embed, query, device_scores, stop_logit, value. One block per batch.
__global__ void k_head(const float* __restrict__ nf, const float* __restrict__ gd,
                       const float* __restrict__ lemb, const float* __restrict__ h,
                       const float* __restrict__ keys,
                       const float* __restrict__ qw, const float* __restrict__ qb,
                       const float* __restrict__ sw1, const float* __restrict__ sb1,
                       const float* __restrict__ sw2, const float* __restrict__ sb2,
                       const float* __restrict__ vw1, const float* __restrict__ vb1,
                       const float* __restrict__ vw2, const float* __restrict__ vb2,
                       float* __restrict__ out) {
  int b = blockIdx.x, t = threadIdx.x;
  __shared__ float wv[N_];
  __shared__ float qi[2 * D_ + 2];
  __shared__ float query[D_];
  __shared__ float red[D_];
  __shared__ float wsum_s;
  if (t < N_) wv[t] = nf[((size_t)(b * N_ + t)) * 11 + 0] * nf[((size_t)(b * N_ + t)) * 11 + 7];
  __syncthreads();
  if (t == 0) {
    float s = 0.f;
    for (int n = 0; n < N_; ++n) s += wv[n];
    wsum_s = fmaxf(s, 1e-8f);
  }
  __syncthreads();
  {
    float s = 0.f;
    for (int n = 0; n < N_; ++n) s += wv[n] * h[((size_t)(b * N_ + n)) * D_ + t];
    qi[t] = s / wsum_s;                 // graph_embed
  }
  qi[D_ + t] = lemb[b * D_ + t];        // layer_embed
  if (t < 2) qi[2 * D_ + t] = gd[b * 2 + t];
  __syncthreads();
  {
    float s = qb[t];
    for (int k = 0; k < 2 * D_ + 2; ++k) s += qi[k] * qw[(size_t)k * D_ + t];
    query[t] = s;
  }
  __syncthreads();
  if (t < N_) {
    float s = 0.f;
    for (int d = 0; d < D_; ++d) s += keys[((size_t)(b * N_ + t)) * D_ + d] * query[d];
    out[b * (N_ + 1) + t] = s * (1.f / 16.f);   // /sqrt(256)
  }
  {
    float s = sb1[t];
    for (int k = 0; k < D_; ++k) s += qi[k] * sw1[(size_t)k * D_ + t];
    s += qi[2 * D_] * sw1[(size_t)D_ * D_ + t] + qi[2 * D_ + 1] * sw1[(size_t)(D_ + 1) * D_ + t];
    red[t] = fmaxf(s, 0.f) * sw2[t];
  }
  __syncthreads();
  for (int s = 128; s > 0; s >>= 1) { if (t < s) red[t] += red[t + s]; __syncthreads(); }
  if (t == 0) out[b * (N_ + 1) + N_] = red[0] + sb2[0];
  __syncthreads();
  {
    float s = vb1[t];
    for (int k = 0; k < 2 * D_ + 2; ++k) s += qi[k] * vw1[(size_t)k * D_ + t];
    red[t] = fmaxf(s, 0.f) * vw2[t];
  }
  __syncthreads();
  for (int s = 128; s > 0; s >>= 1) { if (t < s) red[t] += red[t + s]; __syncthreads(); }
  if (t == 0) out[B_ * (N_ + 1) + b] = red[0] + vb2[0];
}

__global__ void k_copy(const float* __restrict__ src, float* __restrict__ dst, int n) {
  int idx = blockIdx.x * blockDim.x + threadIdx.x;
  if (idx < n) dst[idx] = src[idx];
}

extern "C" void kernel_launch(void* const* d_in, const int* in_sizes, int n_in,
                              void* d_out, int out_size, void* d_ws, size_t ws_size,
                              hipStream_t stream) {
  const float* node_feats = (const float*)d_in[0];
  const float* edge_feats = (const float*)d_in[1];
  const unsigned char* adj = (const unsigned char*)d_in[2];
  const float* layer_costs = (const float*)d_in[3];
  const float* gdyn  = (const float*)d_in[4];
  const float* p_w   = (const float*)d_in[5];
  const float* p_b   = (const float*)d_in[6];
  const float* g_w1  = (const float*)d_in[7];
  const float* g_b1  = (const float*)d_in[8];
  const float* g_w2  = (const float*)d_in[9];
  const float* g_b2  = (const float*)d_in[10];
  const float* u_w1  = (const float*)d_in[11];
  const float* u_b1  = (const float*)d_in[12];
  const float* u_w2  = (const float*)d_in[13];
  const float* u_b2  = (const float*)d_in[14];
  const float* ln_g  = (const float*)d_in[15];
  const float* ln_b  = (const float*)d_in[16];
  const float* conv_w = (const float*)d_in[17];
  const float* conv_b = (const float*)d_in[18];
  const float* lp_w  = (const float*)d_in[19];
  const float* lp_b  = (const float*)d_in[20];
  const float* q_w   = (const float*)d_in[21];
  const float* q_b   = (const float*)d_in[22];
  const float* k_w   = (const float*)d_in[23];
  const float* k_b   = (const float*)d_in[24];
  const float* s_w1  = (const float*)d_in[25];
  const float* s_b1  = (const float*)d_in[26];
  const float* s_w2  = (const float*)d_in[27];
  const float* s_b2  = (const float*)d_in[28];
  const float* v_w1  = (const float*)d_in[29];
  const float* v_b1  = (const float*)d_in[30];
  const float* v_w2  = (const float*)d_in[31];
  const float* v_b2  = (const float*)d_in[32];
  float* out = (float*)d_out;
  (void)in_sizes; (void)n_in; (void)out_size; (void)ws_size;

  char* ws = (char*)d_ws;
  size_t off = 0;
  auto alloc = [&](size_t bytes) -> void* {
    off = (off + 255) & ~(size_t)255;
    void* p = ws + off;
    off += bytes;
    return p;
  };
  const int R = B_ * N_;                                  // 1024 rows
  float*  h    = (float*) alloc((size_t)R * D_ * 4);
  half_t* h16  = (half_t*)alloc((size_t)R * D_ * 2);
  float*  pre1 = (float*) alloc((size_t)R * EH_ * 4);
  float*  aggm = (float*) alloc((size_t)R * D_ * 4);
  float*  aggx = (float*) alloc((size_t)R * D_ * 4);
  half_t* ui16 = (half_t*)alloc((size_t)R * 3 * D_ * 2);
  half_t* u1o  = (half_t*)alloc((size_t)R * D_ * 2);
  float*  u2o  = (float*) alloc((size_t)R * D_ * 4);
  float*  keys = (float*) alloc((size_t)R * D_ * 4);
  float*  lemb = (float*) alloc((size_t)B_ * D_ * 4);
  half_t* w1t  = (half_t*)alloc((size_t)NG_ * EH_ * D_ * 2);     // (N=64,K=256) per layer
  half_t* g2t  = (half_t*)alloc((size_t)NG_ * D_ * EH_ * 2);     // (N=256,K=64)
  half_t* u1t  = (half_t*)alloc((size_t)NG_ * D_ * 3 * D_ * 2);  // (N=256,K=768)
  half_t* u2t  = (half_t*)alloc((size_t)NG_ * D_ * D_ * 2);      // (N=256,K=256)
  half_t* kwt  = (half_t*)alloc((size_t)D_ * D_ * 2);            // (N=256,K=256)

  auto tg = [](int total) { return dim3((unsigned)((total + 255) / 256)); };

  for (int i = 0; i < NG_; ++i) {
    k_transpose_h<<<tg(EH_ * D_), 256, 0, stream>>>(g_w1 + (size_t)i * (D_ + 3) * EH_,
                                                    w1t + (size_t)i * EH_ * D_, D_, EH_);
    k_transpose_h<<<tg(D_ * EH_), 256, 0, stream>>>(g_w2 + (size_t)i * EH_ * D_,
                                                    g2t + (size_t)i * D_ * EH_, EH_, D_);
    k_transpose_h<<<tg(3 * D_ * D_), 256, 0, stream>>>(u_w1 + (size_t)i * 3 * D_ * D_,
                                                       u1t + (size_t)i * D_ * 3 * D_, 3 * D_, D_);
    k_transpose_h<<<tg(D_ * D_), 256, 0, stream>>>(u_w2 + (size_t)i * D_ * D_,
                                                   u2t + (size_t)i * D_ * D_, D_, D_);
  }
  k_transpose_h<<<tg(D_ * D_), 256, 0, stream>>>(k_w, kwt, D_, D_);

  k_layer_embed<<<B_, 256, 0, stream>>>(layer_costs, conv_w, conv_b, lp_w, lp_b, lemb);
  k_node_proj<<<R, 256, 0, stream>>>(node_feats, p_w, p_b, h, h16);

  for (int i = 0; i < NG_; ++i) {
    // per-node part of gate MLP layer 1 (factored out of the edge loop)
    k_gemm<<<dim3(R / 16, EH_ / 16), 32, 0, stream>>>(h16, w1t + (size_t)i * EH_ * D_,
                                                      g_b1 + i * EH_, pre1, (half_t*)nullptr,
                                                      D_, EH_, 0);
    // per-edge gate (the dominant 13 GFLOP) + masked mean/max aggregation
    k_edge<<<R, 256, 0, stream>>>(pre1, g2t + (size_t)i * D_ * EH_, g_b2 + i * D_,
                                  g_w1 + (size_t)i * (D_ + 3) * EH_ + (size_t)D_ * EH_,
                                  edge_feats, adj, h, aggm, aggx);
    k_concat<<<tg(R * 3 * D_), 256, 0, stream>>>(h, aggm, aggx, ui16);
    k_gemm<<<dim3(R / 16, D_ / 16), 32, 0, stream>>>(ui16, u1t + (size_t)i * D_ * 3 * D_,
                                                     u_b1 + i * D_, (float*)nullptr, u1o,
                                                     3 * D_, D_, 1);
    k_gemm<<<dim3(R / 16, D_ / 16), 32, 0, stream>>>(u1o, u2t + (size_t)i * D_ * D_,
                                                     u_b2 + i * D_, u2o, (half_t*)nullptr,
                                                     D_, D_, 0);
    k_ln<<<R, 256, 0, stream>>>(u2o, ln_g + i * D_, ln_b + i * D_, h, h16);
  }
  k_gemm<<<dim3(R / 16, D_ / 16), 32, 0, stream>>>(h16, kwt, k_b, keys, (half_t*)nullptr,
                                                   D_, D_, 0);
  k_head<<<B_, 256, 0, stream>>>(node_feats, gdyn, lemb, h, keys, q_w, q_b,
                                 s_w1, s_b1, s_w2, s_b2, v_w1, v_b1, v_w2, v_b2, out);
  k_copy<<<tg(R * D_), 256, 0, stream>>>(h, out + B_ * (N_ + 1) + B_, R * D_);
}
